// CgpHmmLayer_82944408420951
// MI455X (gfx1250) — matrix operand order repaired and checked by reading, not compiled, not run on hardware
//
#include <hip/hip_runtime.h>
#include <math.h>
#include <stdint.h>

// Problem dims (fixed by reference)
#define BATCH   32
#define TLEN    2048
#define SDIM    512
#define EDIM    32
#define HALF_N  256          // N columns per workgroup (SDIM/2)
#define MROWS   16           // batch rows per group (one WMMA M tile)
#define KT      16           // K tiles of 32 (SDIM/32)
#define NT_WG   16           // N tiles of 16 per WG (HALF_N/16)

typedef __attribute__((ext_vector_type(16))) __bf16 v16bf;
typedef __attribute__((ext_vector_type(8)))  float  v8f;

struct Frag32 { uint4 lo, hi; };   // 32 bytes = one 16-element bf16 operand fragment

__device__ __forceinline__ uint32_t f2bf(float x) {
    uint32_t u = __float_as_uint(x);
    return (u + 0x7FFFu + ((u >> 16) & 1u)) >> 16;   // round-to-nearest-even bf16
}

// ---------------------------------------------------------------------------
// Prep: one-hot [B,T,E] -> obs index [B,T]; also zero the sync flags.
// ---------------------------------------------------------------------------
extern "C" __global__ void __launch_bounds__(256)
hmm_prep(const float* __restrict__ inputs, int* __restrict__ obs,
         unsigned int* __restrict__ flags) {
    int id = blockIdx.x * 256 + threadIdx.x;
    if (id < 4) flags[id] = 0;                 // reset handshake counters each call
    if (id < BATCH * TLEN) {
        const float* p = inputs + (size_t)id * EDIM;
        int best = 0; float bv = p[0];
#pragma unroll
        for (int e = 1; e < EDIM; ++e) { float v = p[e]; if (v > bv) { bv = v; best = e; } }
        obs[id] = best;
    }
}

// ---------------------------------------------------------------------------
// Main: 4 workgroups = 2 independent batch-groups x 2 N-halves.
// Each WG holds its 512x256 bf16 slice of A in LDS, pre-swizzled into
// WMMA B-operand fragments; alpha (16x512 bf16) lives in LDS row-major.
// ---------------------------------------------------------------------------
extern "C" __global__ void __launch_bounds__(256)
hmm_fwd(const float* __restrict__ A, const float* __restrict__ Bem,
        const float* __restrict__ pi, const int* __restrict__ obs,
        float* __restrict__ aEx, float* __restrict__ zEx,
        unsigned int* __restrict__ flags, float* __restrict__ out) {

    __shared__ __align__(16) uint32_t       Ablk[KT * NT_WG * 256];      // 256 KB, swizzled B tiles
    __shared__ __align__(16) unsigned short alphaH[MROWS * 520];         // 16.25 KB (row stride 520 = pad)
    __shared__ __align__(16) float          BemL[HALF_N * EDIM];         // 32 KB emission chunk
    __shared__ float zpart[16];
    __shared__ float zTot[16];
    __shared__ int   obsCur[16];

    const int tid   = threadIdx.x;
    const int lane  = tid & 31;
    const int wv    = tid >> 5;          // 8 waves
    const int hi16  = lane >> 4;         // 0: lanes 0-15, 1: lanes 16-31
    const int lrow  = lane & 15;
    const int group = blockIdx.x >> 1;   // which 16 batches
    const int half  = blockIdx.x & 1;    // which 256 columns of A
    const int mbase = group * MROWS;

    // ---- one-time: emission chunk (rows half*256 .. +256) ----
    for (int i = tid; i < HALF_N * EDIM; i += 256)
        BemL[i] = Bem[half * HALF_N * EDIM + i];

    // ---- one-time: load + bf16-convert + swizzle A slice into B-operand tiles ----
    // Tile (kt,nt) = 32x16 bf16, stored as 1KB block; lane holds 8 packed dwords:
    // lane n (0-15): K = {2v,2v+1} (+8 for v>=4); lanes 16-31: K += 8.  (ISA 7.12.2, 16-bit B)
    for (int bi = wv; bi < KT * NT_WG; bi += 8) {
        int kt = bi >> 4, nt = bi & 15;
        int N  = half * HALF_N + nt * 16 + lrow;
#pragma unroll
        for (int v = 0; v < 8; ++v) {
            int K0 = kt * 32 + 2 * v + (v >= 4 ? 8 : 0) + hi16 * 8;
            float f0 = A[(size_t)K0 * SDIM + N];
            float f1 = A[(size_t)(K0 + 1) * SDIM + N];
            Ablk[bi * 256 + lane * 8 + v] = f2bf(f0) | (f2bf(f1) << 16);
        }
    }

    // ---- t = 0: unnormalized alpha0 = pi * Bem[:,obs0]; z0 folds into step-1 z ----
    if (tid < 16) obsCur[tid] = obs[(mbase + tid) * TLEN + 0];
    __syncthreads();
    for (int i = tid; i < MROWS * SDIM; i += 256) {
        int m = i >> 9, s = i & (SDIM - 1);
        float val = pi[s] * Bem[s * EDIM + obsCur[m]];
        alphaH[m * 520 + s] = (unsigned short)f2bf(val);
    }
    float loglik = 0.0f;
    __syncthreads();

    const int self = group * 2 + half, peer = group * 2 + (1 - half);

    for (int t = 1; t < TLEN; ++t) {
        if (tid < 16)      obsCur[tid] = obs[(mbase + tid) * TLEN + t];
        else if (tid < 32) zpart[tid - 16] = 0.0f;
        __syncthreads();   // alpha/obs/zpart ready

        // ---- GEMM: this wave computes 2 N-tiles of (alpha @ A_half).
        // kt outer so the shared alpha A-fragment is loaded ONCE per K-step
        // and fed to both tiles' WMMAs (96 ds_load_b128/wave/step vs 128). ----
        v8f acc[2];
#pragma unroll
        for (int j = 0; j < 2; ++j)
#pragma unroll
            for (int q = 0; q < 8; ++q) acc[j][q] = 0.0f;

#pragma unroll
        for (int kt = 0; kt < KT; ++kt) {
            // A-operand 16x32 bf16: lane row = lrow, K base kt*32 (+8 for lanes>=16)
            const unsigned short* ap = &alphaH[lrow * 520 + kt * 32 + hi16 * 8];
            Frag32 fa;
            fa.lo = *(const uint4*)ap;
            fa.hi = *(const uint4*)(ap + 16);
            v16bf av = __builtin_bit_cast(v16bf, fa);
#pragma unroll
            for (int j = 0; j < 2; ++j) {
                const int nt = wv * 2 + j;
                // B-operand: pre-swizzled tile, lane-contiguous 32B
                const uint32_t* bp = &Ablk[(kt * 16 + nt) * 256 + lane * 8];
                Frag32 fb;
                fb.lo = *(const uint4*)bp;
                fb.hi = *(const uint4*)(bp + 4);
                v16bf bv = __builtin_bit_cast(v16bf, fb);
                acc[j] = __builtin_amdgcn_wmma_f32_16x16x32_bf16(
                            false, av, false, bv, (short)0, acc[j], false, false);
            }
        }

        // ---- emission multiply + per-row partial sums ----
        float vals[2][8];
#pragma unroll
        for (int j = 0; j < 2; ++j) {
            const int ncol = (wv * 2 + j) * 16 + lrow;   // column within this half
#pragma unroll
            for (int v = 0; v < 8; ++v) {
                const int m = v + hi16 * 8;              // C layout: VGPR v -> row v / v+8
                float val = acc[j][v] * BemL[ncol * EDIM + obsCur[m]];
                vals[j][v] = val;
                float red = val;
#pragma unroll
                for (int off = 8; off >= 1; off >>= 1)
                    red += __shfl_xor(red, off, 32);     // sum across 16-lane half
                if (lrow == 0) atomicAdd(&zpart[m], red);
            }
        }
        __syncthreads();   // zpart complete

        // ---- publish own chunk + partial z; release flag ----
        float* aOwn = aEx + (size_t)(self * 2 + (t & 1)) * (MROWS * HALF_N);
#pragma unroll
        for (int j = 0; j < 2; ++j) {
            const int ncol = (wv * 2 + j) * 16 + lrow;
#pragma unroll
            for (int v = 0; v < 8; ++v)
                aOwn[(v + hi16 * 8) * HALF_N + ncol] = vals[j][v];
        }
        if (tid < 16) zEx[(self * 2 + (t & 1)) * 16 + tid] = zpart[tid];
        __threadfence();
        __syncthreads();
        if (tid == 0)
            __hip_atomic_store(&flags[self], (unsigned)t,
                               __ATOMIC_RELEASE, __HIP_MEMORY_SCOPE_AGENT);
        // ---- wait for partner's step t ----
        if (tid == 0) {
            while (__hip_atomic_load(&flags[peer], __ATOMIC_ACQUIRE,
                                     __HIP_MEMORY_SCOPE_AGENT) < (unsigned)t)
                __builtin_amdgcn_s_sleep(1);
        }
        __syncthreads();
        __threadfence();   // acquire visibility of partner's data for all waves

        // ---- total z, log-likelihood ----
        const float* zP = zEx + (size_t)(peer * 2 + (t & 1)) * 16;
        if (tid < 16) zTot[tid] = zpart[tid] + zP[tid];
        __syncthreads();
        if (tid < 16) loglik += logf(zTot[tid]);

        // ---- build normalized bf16 alpha for next step: own half from regs ----
#pragma unroll
        for (int j = 0; j < 2; ++j) {
            const int ncol = (wv * 2 + j) * 16 + lrow;
#pragma unroll
            for (int v = 0; v < 8; ++v) {
                const int m = v + hi16 * 8;
                float a = vals[j][v] / zTot[m];
                alphaH[m * 520 + half * HALF_N + ncol] = (unsigned short)f2bf(a);
            }
        }
        // ---- partner half from L2 ----
        const float* aP = aEx + (size_t)(peer * 2 + (t & 1)) * (MROWS * HALF_N);
        const int pbase = (1 - half) * HALF_N;
        for (int i = tid; i < MROWS * HALF_N; i += 256) {
            const int m = i >> 8, c = i & 255;
            float a = aP[i] / zTot[m];
            alphaH[m * 520 + pbase + c] = (unsigned short)f2bf(a);
        }
        __syncthreads();   // alpha built; zpart free to be re-zeroed
    }

    if (half == 0 && tid < 16) out[mbase + tid] = loglik;
}

// ---------------------------------------------------------------------------
extern "C" void kernel_launch(void* const* d_in, const int* in_sizes, int n_in,
                              void* d_out, int out_size, void* d_ws, size_t ws_size,
                              hipStream_t stream) {
    const float* inputs = (const float*)d_in[0];   // [32,2048,32] f32 one-hot
    const float* A      = (const float*)d_in[1];   // [512,512]
    const float* Bem    = (const float*)d_in[2];   // [512,32]
    const float* pi     = (const float*)d_in[3];   // [512]

    char* ws = (char*)d_ws;
    int*          obsIdx = (int*)ws;                                   // 256 KB
    float*        aEx    = (float*)(ws + 262144);                      // 4*2*16KB = 128 KB
    float*        zEx    = (float*)(ws + 262144 + 131072);             // 512 B
    unsigned int* flags  = (unsigned int*)(ws + 262144 + 131072 + 512);

    hmm_prep<<<(BATCH * TLEN + 255) / 256, 256, 0, stream>>>(inputs, obsIdx, flags);
    hmm_fwd<<<4, 256, 0, stream>>>(A, Bem, pi, obsIdx, aEx, zEx, flags, (float*)d_out);
}
